// SpiralConvConvBlock_83631603187776
// MI455X (gfx1250) — matrix operand order repaired and checked by reading, not compiled, not run on hardware
//
#include <hip/hip_runtime.h>
#include <cstdint>
#include <cstddef>

// ---------------------------------------------------------------------------
// SpiralConv block for MI455X (gfx1250, wave32):
//   h   = x @ W_in^T + b_in                  (bf16 WMMA GEMM, fp32 accum)
//   w_t = p*w_{t-1} + p0*h_t                 (segmented complex scan)
//   y   = silu(Re(w))                        (fused, in place over h)
//   out = y @ W_out^T + b_out                (bf16 WMMA GEMM, fp32 accum)
// Weights pre-converted to bf16 once; GEMM B-tiles streamed into LDS with
// CDNA5 async-to-LDS (ASYNCcnt) instead of VGPR round-trips.
// ---------------------------------------------------------------------------

#define BM 128
#define BN 128
#define BK 32
#define LDS_STRIDE 40   // 32 + 8 halfs pad -> 80B row stride, conflict-free b128

typedef __attribute__((ext_vector_type(16))) __bf16 v16bf;
typedef __attribute__((ext_vector_type(8)))  float  v8f;

union Frag {
  uint4 q[2];   // 32 bytes
  v16bf v;      // 16 bf16 per lane = A(16x32) or B(32x16) fragment
};

// fp32 -> bf16 round-to-nearest-even
__device__ __forceinline__ unsigned int f2bf1(float f) {
  unsigned int u = __float_as_uint(f);
  return (u + 0x7FFFu + ((u >> 16) & 1u)) >> 16;
}
__device__ __forceinline__ unsigned int pack_bf2(float lo, float hi) {
  return f2bf1(lo) | (f2bf1(hi) << 16);
}

// One-shot fp32 -> bf16 conversion (n % 1024 == 0)
__global__ __launch_bounds__(256)
void f32_to_bf16_kernel(const float* __restrict__ src,
                        unsigned short* __restrict__ dst) {
  const int i = (blockIdx.x * 256 + threadIdx.x) * 4;
  const float4 v = *(const float4*)&src[i];
  *(uint2*)&dst[i] = make_uint2(pack_bf2(v.x, v.y), pack_bf2(v.z, v.w));
}

// C[M,N] = A[M,K](fp32) * W[N,K](bf16)^T + bias[N]
__global__ __launch_bounds__(256)
void gemm_bf16_wmma(const float* __restrict__ A,
                    const unsigned short* __restrict__ W,   // bf16 bits
                    const float* __restrict__ bias,
                    float* __restrict__ C,
                    int M, int N, int K) {
  __shared__ unsigned short Asmem[BM * LDS_STRIDE];
  __shared__ unsigned short Bsmem[BN * LDS_STRIDE];

  const int tid  = threadIdx.x;
  const int lane = tid & 31;
  const int wave = tid >> 5;   // 8 waves
  const int wm   = wave >> 1;  // 0..3 : 32-row slab
  const int wn   = wave & 1;   // 0..1 : 64-col slab
  const int lm   = lane & 15;
  const int lh   = lane >> 4;  // lane half

  const int rowBase = blockIdx.y * BM;
  const int colBase = blockIdx.x * BN;

  v8f acc[2][4];
  #pragma unroll
  for (int i = 0; i < 2; ++i)
    #pragma unroll
    for (int j = 0; j < 4; ++j)
      #pragma unroll
      for (int r = 0; r < 8; ++r) acc[i][j][r] = 0.0f;

  for (int kk = 0; kk < K; kk += BK) {
    __syncthreads();   // previous iteration's LDS reads complete

    // --- B tile: async global->LDS copy of bf16 weights (ASYNCcnt path) ---
    // 512 chunks of 16B; chunk c -> row c/4, 8-half group c%4.
    #pragma unroll
    for (int c = tid; c < BN * 4; c += 256) {
      const int row = c >> 2;
      const int q   = c & 3;
      const unsigned ldsa =
          (unsigned)(uintptr_t)&Bsmem[row * LDS_STRIDE + q * 8];
      const unsigned short* gsrc =
          &W[(size_t)(colBase + row) * K + kk + q * 8];
      asm volatile("global_load_async_to_lds_b128 %0, %1, off"
                   :: "v"(ldsa), "v"(gsrc) : "memory");
    }

    // --- A tile: fp32 global -> bf16 LDS (RNE convert in VALU) ---
    #pragma unroll
    for (int c = tid; c < BM * (BK / 4); c += 256) {
      const int row = c >> 3;
      const int kq  = (c & 7) << 2;
      const float4 va = *(const float4*)&A[(size_t)(rowBase + row) * K + kk + kq];
      *(uint2*)&Asmem[row * LDS_STRIDE + kq] =
          make_uint2(pack_bf2(va.x, va.y), pack_bf2(va.z, va.w));
    }

    asm volatile("s_wait_asynccnt 0x0" ::: "memory");
    __syncthreads();   // whole tile (A via DScnt, B via ASYNCcnt) visible

    // A fragment (ISA 16-bit A 16x32 layout):
    //   lanes 0-15 : K 0..7 in q[0], K 16..23 in q[1]
    //   lanes16-31 : K 8..15 in q[0], K 24..31 in q[1]
    Frag af[2];
    #pragma unroll
    for (int i = 0; i < 2; ++i) {
      const unsigned short* ap = &Asmem[(wm * 32 + i * 16 + lm) * LDS_STRIDE];
      af[i].q[0] = *(const uint4*)(ap + lh * 8);
      af[i].q[1] = *(const uint4*)(ap + 16 + lh * 8);
    }
    // B fragment: lane holds column n = lm; lanes 0-15 K 0..7 / 16..23,
    // lanes 16-31 K 8..15 / 24..31 (same order as A).
    Frag bfr[4];
    #pragma unroll
    for (int j = 0; j < 4; ++j) {
      const unsigned short* bp =
          &Bsmem[(wn * 64 + j * 16 + lm) * LDS_STRIDE + lh * 8];
      bfr[j].q[0] = *(const uint4*)(bp);
      bfr[j].q[1] = *(const uint4*)(bp + 16);
    }

    #pragma unroll
    for (int i = 0; i < 2; ++i)
      #pragma unroll
      for (int j = 0; j < 4; ++j)
        acc[i][j] = __builtin_amdgcn_wmma_f32_16x16x32_bf16(
            false, af[i].v, false, bfr[j].v, (short)0, acc[i][j], false, false);
  }

  // Epilogue: VGPR r holds M=r (lanes 0-15) / M=8+r (lanes 16-31)
  #pragma unroll
  for (int j = 0; j < 4; ++j) {
    const int col = colBase + wn * 64 + j * 16 + lm;
    const float bj = bias[col];
    #pragma unroll
    for (int i = 0; i < 2; ++i) {
      const int row0 = rowBase + wm * 32 + i * 16 + (lh << 3);
      #pragma unroll
      for (int r = 0; r < 8; ++r)
        C[(size_t)(row0 + r) * N + col] = acc[i][j][r] + bj;
    }
  }
}

// ---------------------------------------------------------------------------
// Segmented complex scan: w[t] = p*w[t-1] + p0*h[t], w[-1] = last,
// y[t] = silu(Re(w[t])).  L = NSEG * SEG; per-segment chains of length SEG
// run in parallel; incoming states chained with the closed-form p^SEG.
// ---------------------------------------------------------------------------
#define SEG  256
#define NSEG 8

__device__ __forceinline__ void load_p(const float* phazor, int c,
                                       float& pr, float& pi) {
  const float ar = phazor[2 * c + 0];
  const float ai = phazor[2 * c + 1];
  const float r  = sqrtf(ar * ar + ai * ai);
  const float s  = __expf(-r) / fmaxf(r, 1e-30f);
  pr = ar * s; pi = ai * s;
}

// Phase 1: per-segment accumulation with zero initial state.
// idx = (b*NSEG + s)*DH + c  (c fastest -> coalesced h reads)
__global__ __launch_bounds__(256)
void scan_phase1(const float* __restrict__ h,
                 const float* __restrict__ phazor,
                 const float* __restrict__ phazor_init,
                 float* __restrict__ segA_re, float* __restrict__ segA_im,
                 int L, int DH) {
  const int idx = blockIdx.x * 256 + threadIdx.x;
  const int c   = idx % DH;
  const int bs  = idx / DH;
  const int s   = bs % NSEG;
  const int b   = bs / NSEG;

  float pr, pi;  load_p(phazor, c, pr, pi);
  const float p0r = phazor_init[2 * c + 0];
  const float p0i = phazor_init[2 * c + 1];

  const float* base = h + ((size_t)b * L + (size_t)s * SEG) * DH + c;
  float ar = 0.0f, ai = 0.0f;
  for (int i = 0; i < SEG; ++i) {
    const float v  = base[(size_t)i * DH];
    const float nr = pr * ar - pi * ai + p0r * v;
    const float ni = pr * ai + pi * ar + p0i * v;
    ar = nr; ai = ni;
  }
  segA_re[idx] = ar;
  segA_im[idx] = ai;
}

// Phase 2: serial combine across NSEG segments per channel.
// segW[s] = incoming state of segment s;  w <- p^SEG * w + A[s]
__global__ __launch_bounds__(256)
void scan_phase2(const float* __restrict__ phazor,
                 const float* __restrict__ last_re,
                 const float* __restrict__ last_im,
                 const float* __restrict__ segA_re,
                 const float* __restrict__ segA_im,
                 float* __restrict__ segW_re, float* __restrict__ segW_im,
                 int DH) {
  const int idx = blockIdx.x * 256 + threadIdx.x;   // b*DH + c
  const int c   = idx % DH;
  const int b   = idx / DH;

  float pr, pi;  load_p(phazor, c, pr, pi);
  // p^SEG by repeated squaring (SEG = 2^8)
  float qr = pr, qi = pi;
  #pragma unroll
  for (int k = 0; k < 8; ++k) {
    const float nr = qr * qr - qi * qi;
    const float ni = 2.0f * qr * qi;
    qr = nr; qi = ni;
  }

  float wr = last_re[idx], wi = last_im[idx];
  #pragma unroll
  for (int s = 0; s < NSEG; ++s) {
    const int o = (b * NSEG + s) * DH + c;
    segW_re[o] = wr;  segW_im[o] = wi;
    const float nr = qr * wr - qi * wi + segA_re[o];
    const float ni = qr * wi + qi * wr + segA_im[o];
    wr = nr; wi = ni;
  }
}

// Phase 3: replay each segment with known incoming state; write
// silu(Re(w)) in place over h.
__global__ __launch_bounds__(256)
void scan_phase3(float* __restrict__ h,
                 const float* __restrict__ phazor,
                 const float* __restrict__ phazor_init,
                 const float* __restrict__ segW_re,
                 const float* __restrict__ segW_im,
                 int L, int DH) {
  const int idx = blockIdx.x * 256 + threadIdx.x;
  const int c   = idx % DH;
  const int bs  = idx / DH;
  const int s   = bs % NSEG;
  const int b   = bs / NSEG;

  float pr, pi;  load_p(phazor, c, pr, pi);
  const float p0r = phazor_init[2 * c + 0];
  const float p0i = phazor_init[2 * c + 1];

  float wr = segW_re[idx], wi = segW_im[idx];
  float* base = h + ((size_t)b * L + (size_t)s * SEG) * DH + c;
  for (int i = 0; i < SEG; ++i) {
    const float v  = base[(size_t)i * DH];
    const float nr = pr * wr - pi * wi + p0r * v;
    const float ni = pr * wi + pi * wr + p0i * v;
    wr = nr; wi = ni;
    base[(size_t)i * DH] = nr / (1.0f + __expf(-nr));   // silu(Re(w))
  }
}

extern "C" void kernel_launch(void* const* d_in, const int* in_sizes, int n_in,
                              void* d_out, int out_size, void* d_ws, size_t ws_size,
                              hipStream_t stream) {
  (void)in_sizes; (void)n_in; (void)out_size; (void)ws_size;

  const float* x     = (const float*)d_in[0];
  const float* W_in  = (const float*)d_in[1];
  const float* b_in  = (const float*)d_in[2];
  const float* W_out = (const float*)d_in[3];
  const float* b_out = (const float*)d_in[4];
  const float* phz   = (const float*)d_in[5];
  const float* phz0  = (const float*)d_in[6];
  const float* lre   = (const float*)d_in[7];
  const float* lim   = (const float*)d_in[8];
  float* out = (float*)d_out;

  const int Bsz = 4, L = 2048, D = 1024, DH = 2048;
  const int M = Bsz * L;                 // 8192

  // Workspace layout (bytes)
  char* ws = (char*)d_ws;
  float*          h     = (float*)(ws);                         // 64 MB
  unsigned short* Winb  = (unsigned short*)(ws + 67108864);     //  4 MB
  unsigned short* Woutb = (unsigned short*)(ws + 71303168);     //  4 MB
  float* segA_re = (float*)(ws + 75497472);                     // 256 KB
  float* segA_im = (float*)(ws + 75759616);
  float* segW_re = (float*)(ws + 76021760);
  float* segW_im = (float*)(ws + 76283904);

  // Pre-convert weights to bf16 (once; tiles then async-copy to LDS)
  f32_to_bf16_kernel<<<(DH * D) / 1024, 256, 0, stream>>>(W_in, Winb);
  f32_to_bf16_kernel<<<(D * DH) / 1024, 256, 0, stream>>>(W_out, Woutb);

  // GEMM1: h = x @ W_in^T + b_in   (M=8192, N=2048, K=1024)
  gemm_bf16_wmma<<<dim3(DH / BN, M / BM), 256, 0, stream>>>(x, Winb, b_in, h, M, DH, D);

  // Segmented complex scan + silu (in place over h)
  const int nch = Bsz * DH;
  scan_phase1<<<(nch * NSEG) / 256, 256, 0, stream>>>(h, phz, phz0, segA_re, segA_im, L, DH);
  scan_phase2<<<nch / 256, 256, 0, stream>>>(phz, lre, lim, segA_re, segA_im, segW_re, segW_im, DH);
  scan_phase3<<<(nch * NSEG) / 256, 256, 0, stream>>>(h, phz, phz0, segW_re, segW_im, L, DH);

  // GEMM2: out = y @ W_out^T + b_out  (M=8192, N=1024, K=2048)
  gemm_bf16_wmma<<<dim3(D / BN, M / BM), 256, 0, stream>>>(h, Woutb, b_out, out, M, D, DH);
}
